// ChitEncoder_79070347920038
// MI455X (gfx1250) — compile-verified, matrix operand order checked by reading, not run on hardware
//
#include <hip/hip_runtime.h>
#include <hip/hip_bf16.h>
#include <stdint.h>

// Problem dims (fixed by the reference)
#define Bsz 64
#define Tsz 128
#define Esz 512
#define Usz 1024
#define G3  (3 * Usz)     // 3072
#define TB  (Tsz * Bsz)   // 8192

typedef __attribute__((ext_vector_type(16))) __bf16 v16bf;
typedef __attribute__((ext_vector_type(8)))  __bf16 v8bf;
typedef __attribute__((ext_vector_type(8)))  float  v8f;

// float -> bf16 bits, round-to-nearest-even
static __device__ __forceinline__ unsigned short f2bf(float f) {
    unsigned int u = __float_as_uint(f);
    unsigned int r = u + 0x7fffu + ((u >> 16) & 1u);
    return (unsigned short)(r >> 16);
}

// Per-lane bf16 fragment: two contiguous 16B chunks at K=+0..7 (VGPR0-3) and
// K=+16..23 (VGPR4-7). Caller folded (row*ld + k0 + half*8) into p.
static __device__ __forceinline__ v16bf mk16(const unsigned short* p) {
    v8bf lo = *(const v8bf*)(p);
    v8bf hi = *(const v8bf*)(p + 16);
    v16bf r;
#pragma unroll
    for (int i = 0; i < 8; ++i) { r[i] = lo[i]; r[i + 8] = hi[i]; }
    return r;
}

static __device__ __forceinline__ v8f wmma_bf16(v16bf a, v16bf b, v8f c) {
    return __builtin_amdgcn_wmma_f32_16x16x32_bf16(false, a, false, b,
                                                   (short)0, c, false, false);
}

// Depth-2 software-pipelined 16x64 tile MAC over compile-time K.
// Iteration i's WMMAs consume fragments loaded at iteration i-2, and
// sched_group_barriers pin a [10 VMEM-read][4 WMMA] pattern per iteration so
// the backend emits partial s_wait_loadcnt instead of serializing.
template <int K, int LDB>
static __device__ __forceinline__ void gemm_tile16x64(
    const unsigned short* __restrict__ pa,
    const unsigned short* __restrict__ pb0,
    v8f acc[4])
{
    const unsigned short* pb1 = pb0 + (size_t)16 * LDB;
    const unsigned short* pb2 = pb0 + (size_t)32 * LDB;
    const unsigned short* pb3 = pb0 + (size_t)48 * LDB;

    // Stage 0 (k=0) and stage 1 (k=32) fragments
    v16bf aC  = mk16(pa);
    v16bf b0C = mk16(pb0);
    v16bf b1C = mk16(pb1);
    v16bf b2C = mk16(pb2);
    v16bf b3C = mk16(pb3);
    v16bf aN  = mk16(pa  + 32);
    v16bf b0N = mk16(pb0 + 32);
    v16bf b1N = mk16(pb1 + 32);
    v16bf b2N = mk16(pb2 + 32);
    v16bf b3N = mk16(pb3 + 32);

    __builtin_amdgcn_sched_barrier(0);   // keep prologue loads out of the loop pattern

#pragma unroll
    for (int k0 = 0; k0 < K; k0 += 32) {
        const int kf = k0 + 64;          // fetch distance = 2 iterations
        v16bf aF = aN, b0F = b0N, b1F = b1N, b2F = b2N, b3F = b3N;
        if (kf < K) {
            aF  = mk16(pa  + kf);
            b0F = mk16(pb0 + kf);
            b1F = mk16(pb1 + kf);
            b2F = mk16(pb2 + kf);
            b3F = mk16(pb3 + kf);
        }
        acc[0] = wmma_bf16(aC, b0C, acc[0]);
        acc[1] = wmma_bf16(aC, b1C, acc[1]);
        acc[2] = wmma_bf16(aC, b2C, acc[2]);
        acc[3] = wmma_bf16(aC, b3C, acc[3]);

        aC = aN;  b0C = b0N; b1C = b1N; b2C = b2N; b3C = b3N;
        aN = aF;  b0N = b0F; b1N = b1F; b2N = b2F; b3N = b3F;

        // Desired per-iteration schedule: 10 vmem reads, then 4 wmma.
        __builtin_amdgcn_sched_group_barrier(0x0020, 10, 0); // VMEM read
        __builtin_amdgcn_sched_group_barrier(0x0008, 4, 0);  // MFMA/WMMA
    }
    __builtin_amdgcn_sched_barrier(0);
}

__global__ void k_zero_f32(float* p, int n) {
    for (int i = blockIdx.x * blockDim.x + threadIdx.x; i < n; i += gridDim.x * blockDim.x)
        p[i] = 0.f;
}
__global__ void k_zero_u16(unsigned short* p, int n) {
    for (int i = blockIdx.x * blockDim.x + threadIdx.x; i < n; i += gridDim.x * blockDim.x)
        p[i] = 0;
}

// x[t*B+b][:] = bf16(emb[tokens[b][T-1-t]][:])   (reference reverses tokens first)
__global__ void k_embed(const int* __restrict__ tokens, const float* __restrict__ emb,
                        unsigned short* __restrict__ x) {
    int row = blockIdx.x;            // row = t*B + b
    int t = row >> 6, b = row & 63;
    int tok = tokens[b * Tsz + (Tsz - 1 - t)];
    const float* src = emb + (size_t)tok * Esz;
    unsigned short* dst = x + (size_t)row * Esz;
    int i = threadIdx.x * 4;         // 128 threads * 4 = 512
#pragma unroll
    for (int k = 0; k < 4; ++k) dst[i + k] = f2bf(src[i + k]);
}

// dst[N,K] = bf16(src[K,N])  (transpose + convert; B-matrices stored as B^T)
__global__ void k_trans(const float* __restrict__ src, unsigned short* __restrict__ dst,
                        int K, int N) {
    size_t total = (size_t)K * N;
    for (size_t i = blockIdx.x * blockDim.x + threadIdx.x; i < total;
         i += (size_t)gridDim.x * blockDim.x) {
        size_t n = i / K, k = i % K;
        dst[i] = f2bf(src[k * (size_t)N + n]);
    }
}

// Input projection: xg[d][s*B+b][:] = x_row @ W[d] + bi[d]
// A rows for d=1 are time-reversed so the scan can index xg[d][s] uniformly.
__global__ void __launch_bounds__(256) k_gemm_xg(
    const unsigned short* __restrict__ x,   // [TB, E] bf16
    const unsigned short* __restrict__ Wt,  // [2][G3, E] bf16 (W transposed)
    const float* __restrict__ bif, const float* __restrict__ bib,
    float* __restrict__ xg)                 // [2][TB, G3] f32
{
    const int d = blockIdx.z;
    const unsigned short* Wtd = Wt + (size_t)d * G3 * Esz;
    const float* bi = d ? bib : bif;
    float* out = xg + (size_t)d * TB * G3;

    const int wid  = threadIdx.x >> 5;
    const int lane = threadIdx.x & 31;
    const int r    = lane & 15;
    const int half = lane >> 4;

    const int m0 = blockIdx.y * 128 + wid * 16;   // 8 waves -> 128 rows / block
    const int n0 = blockIdx.x * 64;               // 4 n-tiles / wave

    int m = m0 + r;
    int t = m >> 6, b = m & 63;
    int arow = d ? ((Tsz - 1 - t) * Bsz + b) : m;
    const unsigned short* pa  = x   + (size_t)arow * Esz + half * 8;
    const unsigned short* pb0 = Wtd + (size_t)(n0 + r) * Esz + half * 8;

    v8f acc[4] = {};
    gemm_tile16x64<Esz, Esz>(pa, pb0, acc);

#pragma unroll
    for (int j = 0; j < 4; ++j) {
        int col = n0 + j * 16 + r;
        float bias = bi[col];
#pragma unroll
        for (int e = 0; e < 8; ++e) {
            int row = m0 + half * 8 + e;
            out[(size_t)row * G3 + col] = acc[j][e] + bias;
        }
    }
}

// Per-step recurrent GEMM: hg[d] = h_bf16[d] @ U[d] + bh[d]   ([64,1024]x[1024,3072])
__global__ void __launch_bounds__(256) k_gemm_hg(
    const unsigned short* __restrict__ hbm, // [2][B, U] bf16
    const unsigned short* __restrict__ Ut,  // [2][G3, U] bf16 (U transposed)
    const float* __restrict__ bhf, const float* __restrict__ bhb,
    float* __restrict__ hg)                 // [2][B, G3] f32
{
    const int d = blockIdx.z;
    const unsigned short* A  = hbm + (size_t)d * Bsz * Usz;
    const unsigned short* Bm = Ut  + (size_t)d * G3 * Usz;
    const float* bh = d ? bhb : bhf;
    float* out = hg + (size_t)d * Bsz * G3;

    const int wid  = threadIdx.x >> 5;
    const int lane = threadIdx.x & 31;
    const int r    = lane & 15;
    const int half = lane >> 4;

    const int m0 = (wid & 3) * 16;                       // 4 m-tiles cover M=64
    const int n0 = blockIdx.x * 128 + (wid >> 2) * 64;   // 2 n-groups of 64

    const unsigned short* pa  = A  + (size_t)(m0 + r) * Usz + half * 8;
    const unsigned short* pb0 = Bm + (size_t)(n0 + r) * Usz + half * 8;

    v8f acc[4] = {};
    gemm_tile16x64<Usz, Usz>(pa, pb0, acc);

#pragma unroll
    for (int j = 0; j < 4; ++j) {
        int col = n0 + j * 16 + r;
        float bias = bh[col];
#pragma unroll
        for (int e = 0; e < 8; ++e) {
            int row = m0 + half * 8 + e;
            out[(size_t)row * G3 + col] = acc[j][e] + bias;
        }
    }
}

// Fused GRU gates + state update + output accumulation (Keras reset_after, order z,r,h)
__global__ void k_gates(const float* __restrict__ hg,  // [2][B, G3]
                        const float* __restrict__ xg,  // [2][TB, G3]
                        float* __restrict__ hf,        // [2][B, U] f32 state
                        unsigned short* __restrict__ hb, // [2][B, U] bf16 state
                        float* __restrict__ out,       // [B, T, U]
                        float* __restrict__ hidden,    // [B, 2U]
                        int s)
{
    int d = blockIdx.z;
    int idx = blockIdx.x * blockDim.x + threadIdx.x;   // 0..65535
    int b = idx >> 10, j = idx & 1023;

    const float* hgd = hg + (size_t)d * Bsz * G3 + (size_t)b * G3;
    const float* xgd = xg + (size_t)d * TB * G3 + (size_t)(s * Bsz + b) * G3;

    float xz = xgd[j], xr = xgd[Usz + j], xh = xgd[2 * Usz + j];
    float hz = hgd[j], hr = hgd[Usz + j], hh = hgd[2 * Usz + j];

    float z  = 1.f / (1.f + __expf(-(xz + hz)));
    float rr = 1.f / (1.f + __expf(-(xr + hr)));
    float hc = tanhf(xh + rr * hh);

    size_t hidx = (size_t)d * Bsz * Usz + (size_t)b * Usz + j;
    float hold = hf[hidx];
    float hnew = z * hold + (1.f - z) * hc;
    hf[hidx] = hnew;
    hb[hidx] = f2bf(hnew);

    int tt = d ? (Tsz - 1 - s) : s;   // backward output is time-reversed back
    out[(size_t)b * (Tsz * Usz) + (size_t)tt * Usz + j] += hnew;
    if (s == Tsz - 1)
        hidden[(size_t)b * (2 * Usz) + (size_t)d * Usz + j] = hnew;
}

extern "C" void kernel_launch(void* const* d_in, const int* in_sizes, int n_in,
                              void* d_out, int out_size, void* d_ws, size_t ws_size,
                              hipStream_t stream)
{
    const int*   tokens = (const int*)  d_in[0];
    const float* emb    = (const float*)d_in[1];
    const float* Wf     = (const float*)d_in[2];
    const float* Uf     = (const float*)d_in[3];
    const float* bif    = (const float*)d_in[4];
    const float* bhf    = (const float*)d_in[5];
    const float* Wb     = (const float*)d_in[6];
    const float* Ub     = (const float*)d_in[7];
    const float* bib    = (const float*)d_in[8];
    const float* bhb    = (const float*)d_in[9];
    (void)in_sizes; (void)n_in; (void)ws_size;

    // Workspace carve-out (256B aligned sections)
    char* ws = (char*)d_ws;
    size_t off = 0;
    auto carve = [&](size_t bytes) {
        void* p = ws + off;
        off = (off + bytes + 255) & ~(size_t)255;
        return p;
    };
    unsigned short* x     = (unsigned short*)carve((size_t)TB * Esz * 2);      // bf16 embeddings
    unsigned short* Wt    = (unsigned short*)carve((size_t)2 * G3 * Esz * 2);  // W^T bf16 (2 dirs)
    unsigned short* Ut    = (unsigned short*)carve((size_t)2 * G3 * Usz * 2);  // U^T bf16 (2 dirs)
    float*          xg    = (float*)carve((size_t)2 * TB * G3 * 4);            // input projections
    float*          hgbuf = (float*)carve((size_t)2 * Bsz * G3 * 4);           // per-step h@U+bh
    float*          hf    = (float*)carve((size_t)2 * Bsz * Usz * 4);          // h state f32
    unsigned short* hb    = (unsigned short*)carve((size_t)2 * Bsz * Usz * 2); // h state bf16

    float* out    = (float*)d_out;
    float* hidden = out + (size_t)Bsz * Tsz * Usz;

    // Init (harness poisons buffers)
    k_zero_f32<<<512, 256, 0, stream>>>(out, out_size);
    k_zero_f32<<<64, 256, 0, stream>>>(hf, 2 * Bsz * Usz);
    k_zero_u16<<<64, 256, 0, stream>>>(hb, 2 * Bsz * Usz);

    // Prep: gather + convert + transpose weights
    k_embed<<<TB, 128, 0, stream>>>(tokens, emb, x);
    k_trans<<<512, 256, 0, stream>>>(Wf, Wt,                       Esz, G3);
    k_trans<<<512, 256, 0, stream>>>(Wb, Wt + (size_t)G3 * Esz,    Esz, G3);
    k_trans<<<512, 256, 0, stream>>>(Uf, Ut,                       Usz, G3);
    k_trans<<<512, 256, 0, stream>>>(Ub, Ut + (size_t)G3 * Usz,    Usz, G3);

    // Hoisted input projection for both directions (backward pre-time-reversed)
    k_gemm_xg<<<dim3(G3 / 64, TB / 128, 2), 256, 0, stream>>>(x, Wt, bif, bib, xg);

    // Sequential scan: both directions per step
    for (int s = 0; s < Tsz; ++s) {
        k_gemm_hg<<<dim3(G3 / 128, 1, 2), 256, 0, stream>>>(hb, Ut, bhf, bhb, hgbuf);
        k_gates<<<dim3(256, 1, 2), 256, 0, stream>>>(hgbuf, xg, hf, hb, out, hidden, s);
    }
}
